// UltraSparseModel_32590211842697
// MI455X (gfx1250) — compile-verified, matrix operand order checked by reading, not run on hardware
//
#include <hip/hip_runtime.h>
#include <math.h>

#define B_SZ  16384
#define DIN   256
#define H_SZ  1024
#define K_SZ  16
#define LDK   264            // DIN + 8 halfs pad (row stride 528B, 16B aligned)

typedef __attribute__((ext_vector_type(16))) _Float16 v16h;
typedef __attribute__((ext_vector_type(8)))  _Float16 v8h;
typedef __attribute__((ext_vector_type(8)))  float    v8f;

// fast tanh: tanh(|x|) = (1-u)/(1+u), u = 2^(-2*log2(e)*|x|) in (0,1]; exp2->0 => 1. Saturation-safe.
__device__ __forceinline__ float fast_tanh(float x) {
    float ax = __builtin_fabsf(x);
#if __has_builtin(__builtin_amdgcn_exp2f)
    float u = __builtin_amdgcn_exp2f(ax * -2.8853900817779268f);
#else
    float u = __expf(ax * -2.0f);
#endif
#if __has_builtin(__builtin_amdgcn_rcpf)
    float t = (1.0f - u) * __builtin_amdgcn_rcpf(1.0f + u);
#else
    float t = (1.0f - u) / (1.0f + u);
#endif
    return __builtin_copysignf(t, x);
}

// ---------------------------------------------------------------- zero ws ---
__global__ void zero_ws_kernel(unsigned* __restrict__ p, int n) {
    int i = blockIdx.x * blockDim.x + threadIdx.x;
    if (i < n) p[i] = 0u;
}

// ------------------------------------------------- densify sparse weights ---
// Wh = W^T stored [h][d] as f16. One thread per h: serial RMW handles duplicates.
__global__ void densify_kernel(const int* __restrict__ idx,
                               const float* __restrict__ w1,
                               _Float16* __restrict__ Wh) {
    int h = blockIdx.x * blockDim.x + threadIdx.x;
    if (h >= H_SZ) return;
    _Float16* col = Wh + (size_t)h * DIN;
    #pragma unroll
    for (int k = 0; k < K_SZ; ++k) {
        int d = idx[h * K_SZ + k];
        col[d] = (_Float16)((float)col[d] + w1[h * K_SZ + k]);
    }
}

// ------------------------------------------- fully fused WMMA GEMM + MLP ----
// Each block: 64 rows of x, ALL 1024 hidden units (16 N-chunks of 64).
// out[b] = tanh( sum_h tanh((x@W)[b,h] + b1[h]) * w2[h] + b2 )
__global__ __launch_bounds__(256)
void fused_kernel(const float*    __restrict__ x,
                  const _Float16* __restrict__ Wh,
                  const float*    __restrict__ bias1,
                  const float*    __restrict__ w2v,
                  const float*    __restrict__ b2,
                  float*          __restrict__ out) {
    __shared__ __align__(16) _Float16 As[64][LDK];   // x tile, f16, full K=256
    __shared__ __align__(16) _Float16 Bs[64][LDK];   // W^T chunk, full K=256
    __shared__ float red[64];

    const int tid  = threadIdx.x;
    const int lane = tid & 31;
    const int wave = tid >> 5;
    const int mt   = wave & 3;            // 16-row strip
    const int nt   = wave >> 2;           // 32-col strip within chunk
    const int hr   = lane & 15;
    const int kb   = (lane >> 4) << 3;    // frag K base: 0 / 8
    const int mBase = blockIdx.x * 64;

    const int lr = tid >> 2;              // staging row 0..63
    const int lc = (tid & 3) << 3;        // staging col 0,8,16,24

    // ---- stage A (64 x 256 f32 -> f16 in LDS), init red, prefetch B chunk 0 ----
    {
        const float* xp = x + (size_t)(mBase + lr) * DIN + lc;
        #pragma unroll
        for (int kk = 0; kk < 8; ++kk)
            #pragma unroll
            for (int j = 0; j < 8; ++j)
                As[lr][kk * 32 + lc + j] = (_Float16)xp[kk * 32 + j];
    }
    if (tid < 64) red[tid] = 0.0f;

    v8h breg[8];
    {
        const v8h* gp = (const v8h*)(Wh + (size_t)lr * DIN + lc);
        #pragma unroll
        for (int kk = 0; kk < 8; ++kk) breg[kk] = gp[kk * 4];   // stride 32 halfs
    }
    __syncthreads();                       // As complete

    #pragma unroll
    for (int kk = 0; kk < 8; ++kk)
        *(v8h*)&Bs[lr][kk * 32 + lc] = breg[kk];                // store chunk 0

    // ---- hoist A fragments for full K into registers (reused by all 16 chunks) ----
    // 16-bit A layout: lanes 0-15 hold K 0-7 & 16-23, lanes 16-31 hold K 8-15 & 24-31
    v16h afrag[8];
    #pragma unroll
    for (int kk = 0; kk < 8; ++kk) {
        v8h lo = *(const v8h*)&As[mt * 16 + hr][kk * 32 + kb];
        v8h hi = *(const v8h*)&As[mt * 16 + hr][kk * 32 + kb + 16];
        #pragma unroll
        for (int i = 0; i < 8; ++i) { afrag[kk][i] = lo[i]; afrag[kk][8 + i] = hi[i]; }
    }
    __syncthreads();                       // Bs chunk 0 complete

    float srow[8] = {0.f, 0.f, 0.f, 0.f, 0.f, 0.f, 0.f, 0.f};

    for (int c = 0; c < 16; ++c) {
        // register-prefetch next B chunk; loads stay in flight during WMMAs below
        if (c < 15) {
            const v8h* gp = (const v8h*)(Wh + (size_t)((c + 1) * 64 + lr) * DIN + lc);
            #pragma unroll
            for (int kk = 0; kk < 8; ++kk) breg[kk] = gp[kk * 4];
        }

        v8f acc0 = {}; v8f acc1 = {};
        #pragma unroll
        for (int kk = 0; kk < 8; ++kk) {
            v16h bf0, bf1;
            v8h lo = *(const v8h*)&Bs[nt * 32 + hr][kk * 32 + kb];
            v8h hi = *(const v8h*)&Bs[nt * 32 + hr][kk * 32 + kb + 16];
            #pragma unroll
            for (int i = 0; i < 8; ++i) { bf0[i] = lo[i]; bf0[8 + i] = hi[i]; }
            lo = *(const v8h*)&Bs[nt * 32 + 16 + hr][kk * 32 + kb];
            hi = *(const v8h*)&Bs[nt * 32 + 16 + hr][kk * 32 + kb + 16];
            #pragma unroll
            for (int i = 0; i < 8; ++i) { bf1[i] = lo[i]; bf1[8 + i] = hi[i]; }

            acc0 = __builtin_amdgcn_wmma_f32_16x16x32_f16(false, afrag[kk], false, bf0,
                                                          (short)0, acc0, false, false);
            acc1 = __builtin_amdgcn_wmma_f32_16x16x32_f16(false, afrag[kk], false, bf1,
                                                          (short)0, acc1, false, false);
        }

        // fused epilogue: per-lane running sums (cross-lane reduce deferred to the end)
        int   n0  = c * 64 + nt * 32 + hr;
        float wn0 = w2v[n0],      bn0 = bias1[n0];
        float wn1 = w2v[n0 + 16], bn1 = bias1[n0 + 16];
        #pragma unroll
        for (int i = 0; i < 8; ++i)
            srow[i] += fast_tanh(acc0[i] + bn0) * wn0
                     + fast_tanh(acc1[i] + bn1) * wn1;

        __syncthreads();                   // all waves done reading Bs
        if (c < 15) {
            #pragma unroll
            for (int kk = 0; kk < 8; ++kk)
                *(v8h*)&Bs[lr][kk * 32 + lc] = breg[kk];
            __syncthreads();               // next chunk visible
        }
    }

    // ---- final reduction: 16 lanes x 2 waves(nt) -> red[64] -> tanh -> out ----
    // C layout: VGPR i -> M = i + (lane>=16 ? 8 : 0); N = lane%16
    const int mLocal = mt * 16 + ((lane >> 4) << 3);
    #pragma unroll
    for (int i = 0; i < 8; ++i) {
        float v = srow[i];
        #pragma unroll
        for (int off = 8; off > 0; off >>= 1)
            v += __shfl_xor(v, off, 16);
        if (hr == 0) atomicAdd(&red[mLocal + i], v);   // LDS atomic (ds_add_f32)
    }
    __syncthreads();
    if (tid < 64) out[mBase + tid] = fast_tanh(red[tid] + b2[0]);
}

// ---------------------------------------------------------------------------
extern "C" void kernel_launch(void* const* d_in, const int* in_sizes, int n_in,
                              void* d_out, int out_size, void* d_ws, size_t ws_size,
                              hipStream_t stream) {
    (void)in_sizes; (void)n_in; (void)out_size; (void)ws_size;

    const float* x   = (const float*)d_in[0];
    const int*   idx = (const int*)  d_in[1];
    const float* w1  = (const float*)d_in[2];
    const float* b1  = (const float*)d_in[3];
    const float* w2  = (const float*)d_in[4];
    const float* b2  = (const float*)d_in[5];
    float* out = (float*)d_out;

    // workspace: [ Wh : H*DIN f16 = 512KB ]
    _Float16* Wh = (_Float16*)d_ws;
    const int zeroDwords = H_SZ * DIN / 2;                 // f16 buffer as dwords
    zero_ws_kernel<<<(zeroDwords + 255) / 256, 256, 0, stream>>>((unsigned*)d_ws, zeroDwords);
    densify_kernel<<<H_SZ / 256, 256, 0, stream>>>(idx, w1, Wh);
    fused_kernel<<<B_SZ / 64, 256, 0, stream>>>(x, Wh, b1, w2, b2, out);
}
